// Llama2Attention_68607807586655
// MI455X (gfx1250) — compile-verified, matrix operand order checked by reading, not run on hardware
//
#include <hip/hip_runtime.h>

typedef __bf16 bf16;
typedef __attribute__((ext_vector_type(8)))  __bf16 v8bf;
typedef __attribute__((ext_vector_type(16))) __bf16 v16bf;
typedef __attribute__((ext_vector_type(8)))  float  v8f;
typedef __attribute__((ext_vector_type(4)))  unsigned int u32x4;
typedef __attribute__((ext_vector_type(8)))  int i32x8;
typedef __attribute__((ext_vector_type(4)))  int i32x4;

#define S_LEN  2048
#define DMODEL 2048
#define NH     32
#define NKV    8
#define HD     64
#define KVD    1024
#define QKVN   4096   // packed Q(2048) | K(1024) | V(1024)

#define LDS_ROW 40    // 32 bf16 + 8 bf16 TDM pad (4 DWORDs) per row -> 80B stride
#define BUF_ELEMS (384 * LDS_ROW)   // A(128 rows) + B(256 rows) per stage

// ---------------------------------------------------------------------------
// WMMA helper: D = A(16x32 bf16) * B(32x16 bf16) + C(16x16 f32)
// ---------------------------------------------------------------------------
__device__ __forceinline__ v8f wmma_bf16(v16bf a, v16bf b, v8f c) {
  return __builtin_amdgcn_wmma_f32_16x16x32_bf16(
      false, a, false, b, (short)0, c, false, false);
}

// ---------------------------------------------------------------------------
// 16x32 bf16 fragment loaders (A and B operands share per-lane address math
// for K-contiguous storage): lane L<16 holds row L, K={k0..k0+7, k0+16..k0+23};
// lane L+16 holds row L, K={k0+8..k0+15, k0+24..k0+31}.
// ---------------------------------------------------------------------------
__device__ __forceinline__ v16bf load_frag_g(const bf16* __restrict__ base,
                                             int row0, int ld, int k0, int lane) {
  const int r = lane & 15, h = lane >> 4;
  const bf16* p = base + (size_t)(row0 + r) * (size_t)ld + k0 + h * 8;
  v8bf lo = *(const v8bf*)p;
  v8bf hi = *(const v8bf*)(p + 16);
  v16bf out;
#pragma unroll
  for (int i = 0; i < 8; ++i) { out[i] = lo[i]; out[i + 8] = hi[i]; }
  return out;
}

__device__ __forceinline__ v16bf load_frag_lds(const bf16* base, int row0, int ld,
                                               int lane) {
  const int r = lane & 15, h = lane >> 4;
  const bf16* p = base + (row0 + r) * ld + h * 8;
  v8bf lo = *(const v8bf*)p;
  v8bf hi = *(const v8bf*)(p + 16);
  v16bf out;
#pragma unroll
  for (int i = 0; i < 8; ++i) { out[i] = lo[i]; out[i + 8] = hi[i]; }
  return out;
}

// ---------------------------------------------------------------------------
// TDM: DMA a 2-D bf16 tile (tile_d1 rows x 32 cols) from global into LDS,
// padding 4 DWORDs after every 16 DWORDs (one 64B row) -> LDS_ROW stride.
// D# layout per CDNA5 ISA ch.8 (group0 128b, group1 256b, groups 2/3 zero).
// Toolchain uses the 6-arg builtin form: (u32x4, i32x8, i32x4, i32x4, i32x8, cpol).
// ---------------------------------------------------------------------------
__device__ __forceinline__ void tdm_load_2d(unsigned lds_off, const void* gptr,
                                            int tensor_d0, int tensor_d1,
                                            int tile_d1, int stride0) {
  unsigned long long ga = (unsigned long long)gptr;
  u32x4 g0 = {};
  g0[0] = 1u;                                            // count=1, user mode
  g0[1] = lds_off;                                       // lds_addr [63:32]
  g0[2] = (unsigned)(ga & 0xFFFFFFFFu);                  // global_addr lo
  g0[3] = (unsigned)((ga >> 32) & 0x1FFFFFFu)            // global_addr [56:32]
          | (2u << 30);                                  // type=2 ("image")
  i32x8 g1 = {};
  g1[0] = (1 << 16)                                      // data_size = 2B
          | (1 << 20)                                    // pad_enable
          | (3 << 22)                                    // pad_interval: 16 DW
          | (3 << 25);                                   // pad_amount: 4 DW
  g1[1] = (tensor_d0 & 0xFFFF) << 16;                    // tensor_dim0 lo16
  g1[2] = ((unsigned)tensor_d0 >> 16) | ((tensor_d1 & 0xFFFF) << 16);
  g1[3] = ((unsigned)tensor_d1 >> 16) | (32 << 16);      // tile_dim0 = 32
  g1[4] = tile_d1 & 0xFFFF;                              // tile_dim1 (tile_dim2=0)
  g1[5] = stride0;                                       // tensor_dim0_stride lo32
  g1[6] = 0;
  g1[7] = 0;
  i32x4 g2 = {}, g3 = {};
  i32x8 g4 = {};
  __builtin_amdgcn_tensor_load_to_lds(g0, g1, g2, g3, g4, 0);
}

// ---------------------------------------------------------------------------
// fp32 -> bf16 conversion + weight packing: wqkv = [wq ; wk ; wv] (4096 x 2048)
// ---------------------------------------------------------------------------
__global__ void convert_kernel(const float* __restrict__ x,
                               const float* __restrict__ wq,
                               const float* __restrict__ wk,
                               const float* __restrict__ wv,
                               const float* __restrict__ wo,
                               bf16* __restrict__ xb,
                               bf16* __restrict__ wqkv,
                               bf16* __restrict__ wob) {
  const int stride = gridDim.x * blockDim.x;
  const int tid = blockIdx.x * blockDim.x + threadIdx.x;
  const int n4 = DMODEL * DMODEL;  // 4M
  const int n2 = KVD * DMODEL;     // 2M
  for (int i = tid; i < n4; i += stride) xb[i] = (bf16)x[i];
  for (int i = tid; i < n4; i += stride) wqkv[i] = (bf16)wq[i];
  for (int i = tid; i < n2; i += stride) wqkv[n4 + i] = (bf16)wk[i];
  for (int i = tid; i < n2; i += stride) wqkv[n4 + n2 + i] = (bf16)wv[i];
  for (int i = tid; i < n4; i += stride) wob[i] = (bf16)wo[i];
}

// ---------------------------------------------------------------------------
// C[M,N] = A[M,K] * B[N,K]^T, bf16 in, bf16/f32 out.
// Block = 256 threads (8 waves), tile 128(M) x 256(N), K-step 32.
// TDM double-buffered LDS staging: A tile 128x32, B tile 256x32 per stage.
// ---------------------------------------------------------------------------
template <bool OUT_F32>
__global__ void gemm_kernel(const bf16* __restrict__ A, const bf16* __restrict__ B,
                            void* __restrict__ Cout, int M, int N, int K) {
  __shared__ bf16 smem[2 * BUF_ELEMS];  // 60 KB: 2 x (A 128 + B 256 rows)

  const int lane = threadIdx.x & 31;
  const int wave = threadIdx.x >> 5;
  const int wr = wave >> 2, wc = wave & 3;
  const int m0 = blockIdx.y * 128;
  const int n0 = blockIdx.x * 256;

  // Runtime LDS byte offset of smem (low 32 bits of the generic pointer are
  // the LDS offset per the flat-address aperture rules).
  const unsigned smem_off = (unsigned)(size_t)(void*)&smem[0];

  v8f zero = {};
  v8f acc[4][4];
#pragma unroll
  for (int i = 0; i < 4; ++i)
#pragma unroll
    for (int j = 0; j < 4; ++j) acc[i][j] = zero;

  const int nk = K / 32;
  // Prologue: stage 0 DMA (wave 0 issues; TDM ignores EXEC, one DMA per block)
  if (wave == 0) {
    tdm_load_2d(smem_off, A + (size_t)m0 * K, K, M, 128, K);
    tdm_load_2d(smem_off + 128 * LDS_ROW * 2, B + (size_t)n0 * K, K, N, 256, K);
  }

  for (int ks = 0; ks < nk; ++ks) {
    const int b = ks & 1;
    if (wave == 0) {
      if (ks + 1 < nk) {
        const int kk = (ks + 1) * 32;
        const unsigned nxt = smem_off + (unsigned)((b ^ 1) * BUF_ELEMS * 2);
        tdm_load_2d(nxt, A + (size_t)m0 * K + kk, K, M, 128, K);
        tdm_load_2d(nxt + 128 * LDS_ROW * 2, B + (size_t)n0 * K + kk, K, N, 256, K);
        __builtin_amdgcn_s_wait_tensorcnt(2);  // stage ks retired (in-order)
      } else {
        __builtin_amdgcn_s_wait_tensorcnt(0);
      }
    }
    __syncthreads();  // publish DMA'd tiles to all waves

    const bf16* bufA = smem + b * BUF_ELEMS;
    const bf16* bufB = bufA + 128 * LDS_ROW;
    v16bf af[4], bfm[4];
#pragma unroll
    for (int t = 0; t < 4; ++t)
      af[t] = load_frag_lds(bufA, wr * 64 + 16 * t, LDS_ROW, lane);
#pragma unroll
    for (int t = 0; t < 4; ++t)
      bfm[t] = load_frag_lds(bufB, wc * 64 + 16 * t, LDS_ROW, lane);
#pragma unroll
    for (int i = 0; i < 4; ++i)
#pragma unroll
      for (int j = 0; j < 4; ++j)
        acc[i][j] = wmma_bf16(af[i], bfm[j], acc[i][j]);

    __syncthreads();  // buffer b may be overwritten at stage ks+2
  }

  // C layout: lane holds column n = lane&15, rows e + 8*(lane>>4)
  const int r = lane & 15, h = lane >> 4;
#pragma unroll
  for (int i = 0; i < 4; ++i)
#pragma unroll
    for (int j = 0; j < 4; ++j)
#pragma unroll
      for (int e = 0; e < 8; ++e) {
        const int row = m0 + wr * 64 + 16 * i + e + 8 * h;
        const int col = n0 + wc * 64 + 16 * j + r;
        if (OUT_F32)
          ((float*)Cout)[(size_t)row * N + col] = acc[i][j][e];
        else
          ((bf16*)Cout)[(size_t)row * N + col] = (bf16)acc[i][j][e];
      }
}

// ---------------------------------------------------------------------------
// RoPE on Q,K (in place, bf16) + build V^T so the P*V GEMM is t-contiguous.
// ---------------------------------------------------------------------------
__global__ void rope_kernel(bf16* __restrict__ qkv,
                            const float* __restrict__ fcos,
                            const float* __restrict__ fsin,
                            bf16* __restrict__ vt) {
  const int s = blockIdx.x;
  const int tid = threadIdx.x;
  bf16* row = qkv + (size_t)s * QKVN;
  for (int idx = tid; idx < NH * (HD / 2); idx += blockDim.x) {
    const int head = idx >> 5, p = idx & 31;
    const float c = fcos[s * 32 + p], sn = fsin[s * 32 + p];
    bf16* q = row + head * HD + 2 * p;
    const float re = (float)q[0], im = (float)q[1];
    q[0] = (bf16)(re * c - im * sn);
    q[1] = (bf16)(re * sn + im * c);
  }
  for (int idx = tid; idx < NKV * (HD / 2); idx += blockDim.x) {
    const int head = idx >> 5, p = idx & 31;
    const float c = fcos[s * 32 + p], sn = fsin[s * 32 + p];
    bf16* kp = row + DMODEL + head * HD + 2 * p;
    const float re = (float)kp[0], im = (float)kp[1];
    kp[0] = (bf16)(re * c - im * sn);
    kp[1] = (bf16)(re * sn + im * c);
  }
  for (int d = tid; d < KVD; d += blockDim.x)
    vt[(size_t)d * S_LEN + s] = row[DMODEL + KVD + d];
}

// ---------------------------------------------------------------------------
// Flash-style causal attention. One wave per (16 query rows, head).
// ---------------------------------------------------------------------------
__global__ void attn_kernel(const bf16* __restrict__ qkv,
                            const bf16* __restrict__ vt,
                            bf16* __restrict__ attn) {
  __shared__ bf16 pbuf[16 * 32];
  const int lane = threadIdx.x;
  const int qb = blockIdx.x, h = blockIdx.y, kvh = h >> 2;
  const int q0 = qb * 16;
  const bf16* Qb = qkv + h * HD;                   // ld = QKVN
  const bf16* Kb = qkv + DMODEL + kvh * HD;        // ld = QKVN
  const bf16* Vb = vt + (size_t)kvh * HD * S_LEN;  // rows = d (64), ld = S
  const float scale = 0.125f;                      // 1/sqrt(64)
  const float NEG = -3.0e38f;

  const v16bf qf0 = load_frag_g(Qb, q0, QKVN, 0, lane);
  const v16bf qf1 = load_frag_g(Qb, q0, QKVN, 32, lane);

  v8f o0 = {}, o1 = {}, o2 = {}, o3 = {};
  float mrow[8], lrow[8];
#pragma unroll
  for (int i = 0; i < 8; ++i) { mrow[i] = NEG; lrow[i] = 0.f; }

  const int nloc = lane & 15;
  const int mo = (lane >> 4) * 8;
  const int t_end = q0 + 16;  // causal

  for (int tb = 0; tb < t_end; tb += 32) {
    v8f s0 = {}, s1 = {};
    s0 = wmma_bf16(qf0, load_frag_g(Kb, tb, QKVN, 0, lane), s0);
    s0 = wmma_bf16(qf1, load_frag_g(Kb, tb, QKVN, 32, lane), s0);
    s1 = wmma_bf16(qf0, load_frag_g(Kb, tb + 16, QKVN, 0, lane), s1);
    s1 = wmma_bf16(qf1, load_frag_g(Kb, tb + 16, QKVN, 32, lane), s1);

    float p0[8], p1[8];
#pragma unroll
    for (int i = 0; i < 8; ++i) {
      const int m = q0 + mo + i;
      float v0 = (tb + nloc) <= m ? s0[i] * scale : NEG;
      float v1 = (tb + 16 + nloc) <= m ? s1[i] * scale : NEG;
      float bmax = fmaxf(v0, v1);
#pragma unroll
      for (int d = 8; d >= 1; d >>= 1) bmax = fmaxf(bmax, __shfl_xor(bmax, d, 32));
      const float newm = fmaxf(mrow[i], bmax);
      const float corr = __expf(mrow[i] - newm);
      const float e0 = __expf(v0 - newm);
      const float e1 = __expf(v1 - newm);
      float rs = e0 + e1;
#pragma unroll
      for (int d = 8; d >= 1; d >>= 1) rs += __shfl_xor(rs, d, 32);
      lrow[i] = lrow[i] * corr + rs;
      mrow[i] = newm;
      o0[i] *= corr; o1[i] *= corr; o2[i] *= corr; o3[i] *= corr;
      p0[i] = e0; p1[i] = e1;
    }

    // C-layout (lane = column) -> LDS row-major P[16][32]
#pragma unroll
    for (int i = 0; i < 8; ++i) {
      pbuf[(mo + i) * 32 + nloc] = (bf16)p0[i];
      pbuf[(mo + i) * 32 + nloc + 16] = (bf16)p1[i];
    }
    asm volatile("s_wait_dscnt 0" ::: "memory");  // single-wave LDS handoff
    const v16bf pf = load_frag_lds(pbuf, 0, 32, lane);
    asm volatile("" ::: "memory");

    o0 = wmma_bf16(pf, load_frag_g(Vb, 0, S_LEN, tb, lane), o0);
    o1 = wmma_bf16(pf, load_frag_g(Vb, 16, S_LEN, tb, lane), o1);
    o2 = wmma_bf16(pf, load_frag_g(Vb, 32, S_LEN, tb, lane), o2);
    o3 = wmma_bf16(pf, load_frag_g(Vb, 48, S_LEN, tb, lane), o3);
  }

  bf16* orow = attn + h * HD;
#pragma unroll
  for (int i = 0; i < 8; ++i) {
    const float inv = 1.0f / lrow[i];
    const size_t base = (size_t)(q0 + mo + i) * DMODEL;
    orow[base + 0 * 16 + nloc] = (bf16)(o0[i] * inv);
    orow[base + 1 * 16 + nloc] = (bf16)(o1[i] * inv);
    orow[base + 2 * 16 + nloc] = (bf16)(o2[i] * inv);
    orow[base + 3 * 16 + nloc] = (bf16)(o3[i] * inv);
  }
}

// ---------------------------------------------------------------------------
// Launcher
// ---------------------------------------------------------------------------
extern "C" void kernel_launch(void* const* d_in, const int* in_sizes, int n_in,
                              void* d_out, int out_size, void* d_ws, size_t ws_size,
                              hipStream_t stream) {
  (void)in_sizes; (void)n_in; (void)out_size; (void)ws_size;
  const float* x  = (const float*)d_in[0];
  const float* fc = (const float*)d_in[1];
  const float* fs = (const float*)d_in[2];
  // d_in[3] = mask (recomputed analytically), d_in[8] = start_pos (0)
  const float* wq = (const float*)d_in[4];
  const float* wk = (const float*)d_in[5];
  const float* wv = (const float*)d_in[6];
  const float* wo = (const float*)d_in[7];

  char* ws = (char*)d_ws;
  bf16* xb   = (bf16*)(ws + ((size_t)0 << 20));   //  8 MB
  bf16* wqkv = (bf16*)(ws + ((size_t)8 << 20));   // 16 MB
  bf16* wob  = (bf16*)(ws + ((size_t)24 << 20));  //  8 MB
  bf16* qkv  = (bf16*)(ws + ((size_t)32 << 20));  // 16 MB
  bf16* vtb  = (bf16*)(ws + ((size_t)48 << 20));  //  4 MB
  bf16* attn = (bf16*)(ws + ((size_t)52 << 20));  //  8 MB

  convert_kernel<<<2048, 256, 0, stream>>>(x, wq, wk, wv, wo, xb, wqkv, wob);
  gemm_kernel<false><<<dim3(QKVN / 256, S_LEN / 128), 256, 0, stream>>>(
      xb, wqkv, qkv, S_LEN, QKVN, DMODEL);
  rope_kernel<<<S_LEN, 256, 0, stream>>>(qkv, fc, fs, vtb);
  attn_kernel<<<dim3(S_LEN / 16, NH), 32, 0, stream>>>(qkv, vtb, attn);
  gemm_kernel<true><<<dim3(DMODEL / 256, S_LEN / 128), 256, 0, stream>>>(
      attn, wob, (float*)d_out, S_LEN, DMODEL, DMODEL);
}
